// Transformer_67748814127473
// MI455X (gfx1250) — matrix operand order repaired and checked by reading, not compiled, not run on hardware
//
#include <hip/hip_runtime.h>
#include <hip/hip_bf16.h>

typedef __attribute__((ext_vector_type(16))) __bf16 v16bf;
typedef __attribute__((ext_vector_type(8)))  __bf16 v8bf;
typedef __attribute__((ext_vector_type(8)))  float  v8f;

#define WMMA_BF16(a, b, c) \
  __builtin_amdgcn_wmma_f32_16x16x32_bf16(false, (a), false, (b), (short)0, (c), false, false)

static __device__ __forceinline__ v16bf cat16(v8bf a, v8bf b) {
  v16bf r;
#pragma unroll
  for (int i = 0; i < 8; ++i) { r[i] = a[i]; r[i + 8] = b[i]; }
  return r;
}

// 16x32 bf16 A-operand from a bf16 row-major matrix row.
// Lane l: M = l&15, kh = l>>4; elems 0..15 -> k = k0+kh*16+i (contig 32B).
static __device__ __forceinline__ v16bf load_a_bf16(const __bf16* row, int k0, int lh) {
  const __bf16* p = row + k0 + lh * 16;
  v8bf a0 = *(const v8bf*)(p);
  v8bf a1 = *(const v8bf*)(p + 8);
  return cat16(a0, a1);
}

// 32x16 bf16 B-operand; per lane the 16 K-values are contiguous.
static __device__ __forceinline__ v16bf load_b16(const __bf16* p) {
  v8bf b0 = *(const v8bf*)(p);
  v8bf b1 = *(const v8bf*)(p + 8);
  return cat16(b0, b1);
}

// ---------------- generic wave-tile WMMA GEMM ----------------
// C[M,Nn] = epilogue( A[M,K](bf16 rm) @ Bt[Nn,K](bf16 rm = B^T) )
__global__ __launch_bounds__(256) void gemm_wmma(
    const __bf16* __restrict__ A, const __bf16* __restrict__ Bt,
    const float* __restrict__ bias, const float* __restrict__ resid,
    float* __restrict__ Cf, __bf16* __restrict__ Cbf_rm, __bf16* __restrict__ Cbf_tr,
    int M, int Nn, int K, int do_relu) {
  int lane = threadIdx.x & 31, warp = threadIdx.x >> 5;
  int tile = blockIdx.x * 8 + warp;
  int nt16 = Nn >> 4;
  int mt = tile / nt16, nt = tile - mt * nt16;
  if (mt >= (M >> 4)) return;            // wave-uniform exit (EXEC stays all-1s)
  int l15 = lane & 15, lh = lane >> 4;
  const __bf16* arow = A  + (size_t)(mt * 16 + l15) * K;
  const __bf16* brow = Bt + (size_t)(nt * 16 + l15) * K;
  v8f acc = {};
  for (int k0 = 0; k0 < K; k0 += 32) {
    v16bf a = load_a_bf16(arow, k0, lh);
    v16bf b = load_b16(brow + k0 + lh * 16);
    acc = WMMA_BF16(a, b, acc);
  }
  int n = nt * 16 + l15;
  float bv = bias ? bias[n] : 0.f;
#pragma unroll
  for (int r = 0; r < 8; ++r) {
    int m = mt * 16 + r + lh * 8;
    float v = acc[r] + bv;
    if (do_relu) v = fmaxf(v, 0.f);
    if (resid)  v += resid[(size_t)m * Nn + n];
    if (Cf)     Cf[(size_t)m * Nn + n] = v;
    if (Cbf_rm) Cbf_rm[(size_t)m * Nn + n] = (__bf16)v;
    if (Cbf_tr) Cbf_tr[(size_t)n * M + m] = (__bf16)v;
  }
}

// ---------------- fused relu(QK^T)*mask @ V attention ----------------
// One wave per 16-row block; out tile = 16 x 256 held in 16 f32 accumulators.
__global__ __launch_bounds__(256) void attn_kernel(
    const __bf16* __restrict__ qbf, const __bf16* __restrict__ kbf,
    const __bf16* __restrict__ vT, float* __restrict__ out) {
  __shared__ __bf16 sbuf[8][16 * 32];
  const int E = 256, NS = 2048, BN = 8 * 2048;
  int lane = threadIdx.x & 31, warp = threadIdx.x >> 5;
  int gw = blockIdx.x * 8 + warp;        // 0..1023
  int b = gw >> 7, rb = gw & 127;
  int l15 = lane & 15, lh = lane >> 4;
  __bf16* my = &sbuf[warp][0];

  // cache Q A-operands (rows rb*16..rb*16+15, all 8 k-steps)
  v16bf qa[8];
  const __bf16* qrow = qbf + (size_t)(b * NS + rb * 16 + l15) * E;
#pragma unroll
  for (int ks = 0; ks < 8; ++ks) qa[ks] = load_a_bf16(qrow, ks * 32, lh);

  v8f acc[16];
#pragma unroll
  for (int i = 0; i < 16; ++i) acc[i] = (v8f){};

  int imax  = rb * 16 + 15;
  int jcmax = (imax >> 5) + 1;           // 32-wide j-chunks needed
  for (int jc = 0; jc < jcmax; ++jc) {
#pragma unroll
    for (int half = 0; half < 2; ++half) {
      int jt = jc * 2 + half;
      v8f s = {};
      // B-operand for QK^T is K row-major bf16: contiguous in d per lane
      const __bf16* kb = kbf + (size_t)(b * NS + jt * 16 + l15) * E + lh * 16;
#pragma unroll
      for (int ks = 0; ks < 8; ++ks) s = WMMA_BF16(qa[ks], load_b16(kb + ks * 32), s);
      // relu + causal 1/(i+1) scale, stash as bf16 into wave-private LDS
      int jg = jt * 16 + l15;
#pragma unroll
      for (int r = 0; r < 8; ++r) {
        int ig = rb * 16 + r + lh * 8;
        float v = s[r];
        v = (jg <= ig) ? (fmaxf(v, 0.f) / (float)(ig + 1)) : 0.f;
        my[(r + lh * 8) * 32 + half * 16 + l15] = (__bf16)v;
      }
    }
    asm volatile("s_wait_dscnt 0x0" ::: "memory");
    // re-read as 16x32 bf16 A-operand (transposed lane layout)
    v16bf sa;
    {
      const __bf16* rowp = my + l15 * 32 + lh * 16;
      v8bf s0 = *(const v8bf*)(rowp);
      v8bf s1 = *(const v8bf*)(rowp + 8);
      sa = cat16(s0, s1);
    }
    // S-chunk @ V : 16 n-tiles, B-operand from V^T (contiguous in j per lane)
#pragma unroll
    for (int nt = 0; nt < 16; ++nt) {
      const __bf16* vp = vT + (size_t)(nt * 16 + l15) * BN + b * NS + jc * 32 + lh * 16;
      acc[nt] = WMMA_BF16(sa, load_b16(vp), acc[nt]);
    }
  }
#pragma unroll
  for (int nt = 0; nt < 16; ++nt)
#pragma unroll
    for (int r = 0; r < 8; ++r) {
      int ig = rb * 16 + r + lh * 8;
      out[(size_t)(b * NS + ig) * E + nt * 16 + l15] = acc[nt][r];
    }
}

// ---------------- layernorm: out{f32,bf16} = LN(x [+ add]) * g + b ----------------
__global__ __launch_bounds__(256) void ln_kernel(
    const float* __restrict__ x, const float* __restrict__ add,
    float* __restrict__ out, __bf16* __restrict__ out_bf,
    const float* __restrict__ g, const float* __restrict__ bb) {
  __shared__ float red[256];
  int t = threadIdx.x;
  size_t base = (size_t)blockIdx.x * 256;
  float v = x[base + t];
  if (add) v += add[base + t];
  red[t] = v; __syncthreads();
  for (int s = 128; s > 0; s >>= 1) { if (t < s) red[t] += red[t + s]; __syncthreads(); }
  float mean = red[0] * (1.f / 256.f);
  __syncthreads();
  float d = v - mean;
  red[t] = d * d; __syncthreads();
  for (int s = 128; s > 0; s >>= 1) { if (t < s) red[t] += red[t + s]; __syncthreads(); }
  float r = rsqrtf(red[0] * (1.f / 256.f) + 1e-5f);
  float o = d * r * g[t] + bb[t];
  out[base + t] = o;
  out_bf[base + t] = (__bf16)o;
}

// ---------------- structured embedding ----------------
__global__ void embed_base_kernel(const float* __restrict__ aset,
                                  const float* __restrict__ W, float* __restrict__ base) {
  int idx = blockIdx.x * blockDim.x + threadIdx.x;   // 8*256
  if (idx >= 8 * 256) return;
  int b = idx >> 8, e = idx & 255;
  const float* a = aset + b * 640;
  float s = 0.f;
  for (int i = 0; i < 640; ++i) s += a[i] * W[i * 256 + e];
  base[idx] = s;
}

__global__ void embed_main_kernel(const float* __restrict__ base,
                                  const float* __restrict__ ca, const float* __restrict__ cr,
                                  const float* __restrict__ W,  const float* __restrict__ be,
                                  const float* __restrict__ wpe,
                                  float* __restrict__ H, __bf16* __restrict__ Hbf) {
  int idx = blockIdx.x * blockDim.x + threadIdx.x;   // 8*2048*256
  if (idx >= 8 * 2048 * 256) return;
  int e = idx & 255, n = (idx >> 8) & 2047, b = idx >> 19;
  float s;
  if ((n & 1) == 0) {
    s = base[b * 256 + e];
  } else {
    int t = n >> 1;
    const float* cap = ca + (b * 1024 + t) * 10;
    s = 0.f;
#pragma unroll
    for (int a = 0; a < 10; ++a) s += cap[a] * W[(640 + a) * 256 + e];
    s += cr[b * 1024 + t] * W[650 * 256 + e];
  }
  s += W[651 * 256 + e];                  // "one" feature
  s += (float)(n + 1) * W[652 * 256 + e]; // position feature
  s += be[e] + wpe[n * 256 + e];
  H[idx] = s;
  Hbf[idx] = (__bf16)s;
}

// ---------------- weight transpose + bf16 convert ----------------
__global__ void transpose_to_bf16(const float* __restrict__ W, __bf16* __restrict__ out,
                                  int K, int Nn) {
  int idx = blockIdx.x * blockDim.x + threadIdx.x;
  if (idx >= K * Nn) return;
  int n = idx / K, k = idx - n * K;      // out[n*K + k] = W[k, n]
  out[idx] = (__bf16)W[(size_t)k * Nn + n];
}

// ---------------- final 10-wide projection on even rows ----------------
__global__ void pred_kernel(const float* __restrict__ H, const float* __restrict__ Wp,
                            const float* __restrict__ bp, float* __restrict__ out) {
  int idx = blockIdx.x * blockDim.x + threadIdx.x;   // 8*1024*10
  if (idx >= 8 * 1024 * 10) return;
  int a = idx % 10, t = (idx / 10) & 1023, b = idx / 10240;
  const float* h = H + ((size_t)b * 2048 + 2 * t) * 256;
  float s = bp[a];
  for (int d = 0; d < 256; ++d) s += h[d] * Wp[d * 10 + a];
  out[idx] = s;
}

extern "C" void kernel_launch(void* const* d_in, const int* in_sizes, int n_in,
                              void* d_out, int out_size, void* d_ws, size_t ws_size,
                              hipStream_t stream) {
  (void)in_sizes; (void)n_in; (void)out_size; (void)ws_size;
  const float* action_set = (const float*)d_in[1];
  const float* ca      = (const float*)d_in[2];
  const float* cr      = (const float*)d_in[3];
  const float* W_embed = (const float*)d_in[4];
  const float* b_embed = (const float*)d_in[5];
  const float* wpe     = (const float*)d_in[6];
  const float* Wq      = (const float*)d_in[7];
  const float* Wk      = (const float*)d_in[8];
  const float* Wv      = (const float*)d_in[9];
  const float* ln1g    = (const float*)d_in[10];
  const float* ln1b    = (const float*)d_in[11];
  const float* W1      = (const float*)d_in[12];
  const float* b1      = (const float*)d_in[13];
  const float* W2      = (const float*)d_in[14];
  const float* b2      = (const float*)d_in[15];
  const float* ln2g    = (const float*)d_in[16];
  const float* ln2b    = (const float*)d_in[17];
  const float* W_pred  = (const float*)d_in[18];
  const float* b_pred  = (const float*)d_in[19];
  float* out = (float*)d_out;

  const int BN = 8 * 2048, E = 256;
  char* ws = (char*)d_ws;
  size_t off = 0;
  auto alloc = [&](size_t bytes) -> char* {
    char* p = ws + off;
    off = (off + bytes + 255) & ~(size_t)255;
    return p;
  };
  float*  Hf    = (float*)alloc((size_t)BN * E * 4);
  float*  aux   = (float*)alloc((size_t)BN * E * 4);
  __bf16* Hbf   = (__bf16*)alloc((size_t)BN * E * 2);
  __bf16* qbf   = (__bf16*)alloc((size_t)BN * E * 2);
  __bf16* kbf   = (__bf16*)alloc((size_t)BN * E * 2);
  __bf16* vT    = (__bf16*)alloc((size_t)BN * E * 2);
  __bf16* tmpbf = (__bf16*)alloc((size_t)BN * E * 2);
  float*  base  = (float*)alloc(8 * 256 * 4);
  __bf16* Wt[4][5];
  for (int l = 0; l < 4; ++l)
    for (int m = 0; m < 5; ++m) Wt[l][m] = (__bf16*)alloc((size_t)65536 * 2);

  // one-time (per launch) weight transpose/convert
  for (int l = 0; l < 4; ++l) {
    const float* srcs[5] = {Wq + l * 65536, Wk + l * 65536, Wv + l * 65536,
                            W1 + l * 65536, W2 + l * 65536};
    for (int m = 0; m < 5; ++m)
      transpose_to_bf16<<<65536 / 256, 256, 0, stream>>>(srcs[m], Wt[l][m], 256, 256);
  }

  embed_base_kernel<<<8, 256, 0, stream>>>(action_set, W_embed, base);
  embed_main_kernel<<<BN * E / 256, 256, 0, stream>>>(base, ca, cr, W_embed, b_embed, wpe,
                                                      Hf, Hbf);

  const int gblocks = (BN / 16) * (E / 16) / 8;       // 2048 blocks, 8 waves each
  for (int l = 0; l < 4; ++l) {
    gemm_wmma<<<gblocks, 256, 0, stream>>>(Hbf, Wt[l][0], nullptr, nullptr,
                                           nullptr, qbf, nullptr, BN, E, E, 0);
    gemm_wmma<<<gblocks, 256, 0, stream>>>(Hbf, Wt[l][1], nullptr, nullptr,
                                           nullptr, kbf, nullptr, BN, E, E, 0);
    gemm_wmma<<<gblocks, 256, 0, stream>>>(Hbf, Wt[l][2], nullptr, nullptr,
                                           nullptr, nullptr, vT, BN, E, E, 0);
    attn_kernel<<<128, 256, 0, stream>>>(qbf, kbf, vT, aux);
    ln_kernel<<<BN, 256, 0, stream>>>(Hf, aux, Hf, Hbf, ln1g + l * 256, ln1b + l * 256);
    gemm_wmma<<<gblocks, 256, 0, stream>>>(Hbf, Wt[l][3], b1 + l * 256, nullptr,
                                           nullptr, tmpbf, nullptr, BN, E, E, 1);
    gemm_wmma<<<gblocks, 256, 0, stream>>>(tmpbf, Wt[l][4], b2 + l * 256, Hf,
                                           aux, nullptr, nullptr, BN, E, E, 0);
    ln_kernel<<<BN, 256, 0, stream>>>(aux, nullptr, Hf, Hbf, ln2g + l * 256, ln2b + l * 256);
  }
  pred_kernel<<<(8 * 1024 * 10 + 255) / 256, 256, 0, stream>>>(Hf, W_pred, b_pred, out);
}
